// TextGAT_76218489635505
// MI455X (gfx1250) — compile-verified, matrix operand order checked by reading, not run on hardware
//
#include <hip/hip_runtime.h>
#include <hip/hip_fp16.h>
#include <stdint.h>

typedef __attribute__((ext_vector_type(16))) _Float16 v16h;
typedef __attribute__((ext_vector_type(8)))  float    v8f;

#define KDIM 256      // inner dim of both GEMMs
#define HEADS 2
#define HD 128
#define COUT 4

// ---------------- order-preserving float <-> uint for atomicMax ----------------
static __device__ __forceinline__ unsigned encf(float f) {
    unsigned u = __float_as_uint(f);
    return (u & 0x80000000u) ? ~u : (u | 0x80000000u);
}
static __device__ __forceinline__ float decf(unsigned u) {
    return __uint_as_float((u & 0x80000000u) ? (u ^ 0x80000000u) : ~u);
}
#define ENC_NEG_INF 0x007FFFFFu   // encf(-inf)

// ---------------- weight conversion: W1[k][n] -> f16 Bt1[n][k]; W2[k][c] -> f16 Bt2[n][k] (pad to 16 cols)
__global__ void k_convert_w(const float* __restrict__ W1, const float* __restrict__ W2,
                            unsigned short* __restrict__ Bt1, unsigned short* __restrict__ Bt2)
{
    int i = blockIdx.x * blockDim.x + threadIdx.x;
    if (i < 256 * 256) {
        int n = i / 256, k = i % 256;
        _Float16 v = (_Float16)W1[k * 256 + n];
        Bt1[n * 256 + k] = *(unsigned short*)&v;
    } else if (i < 256 * 256 + 16 * 256) {
        int j = i - 256 * 256;
        int n = j / 256, k = j % 256;
        float f = (n < COUT) ? W2[k * COUT + n] : 0.0f;
        _Float16 v = (_Float16)f;
        Bt2[n * 256 + k] = *(unsigned short*)&v;
    }
}

// ---------------- init accumulators ----------------
__global__ void k_init(int N, float* out1, unsigned* amax1, float* den1,
                       unsigned* amax2, float* den2, float* out_final, const float* b2)
{
    long long i = (long long)blockIdx.x * blockDim.x + threadIdx.x;
    long long tot = (long long)N * 256;
    if (i < tot)      out1[i] = 0.0f;
    if (i < (long long)N * HEADS) { amax1[i] = ENC_NEG_INF; den1[i] = 0.0f; }
    if (i < N)        { amax2[i] = ENC_NEG_INF; den2[i] = 0.0f; }
    if (i < (long long)N * COUT) out_final[i] = b2[i & 3];
}

// ---------------- WMMA GEMM: D[M][ldD] = A[M][256](f32) x Bt(f16 [Ncols][256]) ----------------
// block = 256 threads (8 waves). wave w handles colTile = blockIdx.y*8 + w, rowTile = blockIdx.x.
// Full K stripe (16x256 f16, 8KB) is staged in LDS once; row stride 132 dwords keeps the
// 16-row fragment gather bank-conflict free AND 16B-aligned so A loads are ds_load_b128.
#define LDS_STRIDE 132
__global__ void k_gemm_wmma(const float* __restrict__ A, const unsigned* __restrict__ Bt32,
                            float* __restrict__ D, int M, int nColTiles, int nOutCols, int ldD)
{
    __shared__ __align__(16) unsigned ldsA[16 * LDS_STRIDE];
    const int wave = threadIdx.x >> 5;
    const int lane = threadIdx.x & 31;
    const int row0 = blockIdx.x * 16;
    const int colTile = blockIdx.y * 8 + wave;
    const bool active = (colTile < nColTiles);   // wave-uniform -> EXEC stays all-1s for WMMA
    const int nb   = colTile * 16;
    const int nloc = lane & 15;
    const int mloc = lane & 15;
    const int kgrpP = (lane & 16) ? 4 : 0;       // pair-index offset for the lane half

    // -------- stage the whole 16x256 f32 stripe as f16 pairs (8 float2 per thread) --------
    {
        int m   = threadIdx.x >> 4;     // 0..15
        int kp0 = threadIdx.x & 15;     // pair index seed
        bool rok = (row0 + m < M);
        const float* ap = A + (size_t)(row0 + m) * KDIM;
#pragma unroll
        for (int j = 0; j < 8; ++j) {
            int kp = kp0 + 16 * j;      // pair index 0..127
            float a0 = 0.f, a1 = 0.f;
            if (rok) { a0 = ap[kp * 2]; a1 = ap[kp * 2 + 1]; }
            _Float16 f0 = (_Float16)a0, f1 = (_Float16)a1;
            ldsA[m * LDS_STRIDE + kp] = (unsigned)(*(unsigned short*)&f0) |
                                        ((unsigned)(*(unsigned short*)&f1) << 16);
        }
    }
    __syncthreads();

    v8f acc = {};
    if (active) {
        const unsigned* arow = ldsA + mloc * LDS_STRIDE;
        const unsigned* brow = Bt32 + (size_t)(nb + nloc) * (KDIM / 2);
#pragma unroll
        for (int k0 = 0; k0 < KDIM; k0 += 32) {
            int pb = (k0 >> 1) + kgrpP;          // pair-index base, multiple of 4
            union { v16h v; uint4 q[2]; } fa, fb;
            fa.q[0] = *(const uint4*)&arow[pb];
            fa.q[1] = *(const uint4*)&arow[pb + 8];
            fb.q[0] = *(const uint4*)&brow[pb];
            fb.q[1] = *(const uint4*)&brow[pb + 8];
            acc = __builtin_amdgcn_wmma_f32_16x16x32_f16(false, fa.v, false, fb.v,
                                                         (short)0, acc, false, false);
        }
        // -------- store 16x16 tile --------
        int col = nb + nloc;
        if (col < nOutCols) {
            int mh = (lane >> 4) * 8;
            float* dp = D + (size_t)(row0 + mh) * ldD + col;
            if (row0 + 16 <= M) {
#pragma unroll
                for (int r = 0; r < 8; ++r) dp[(size_t)r * ldD] = acc[r];
            } else {
#pragma unroll
                for (int r = 0; r < 8; ++r)
                    if (row0 + mh + r < M) dp[(size_t)r * ldD] = acc[r];
            }
        }
    }
}

// ---------------- per-node attention logits, layer 1 ----------------
__global__ void k_logits1(int N, const float* __restrict__ h, const float* __restrict__ a_src,
                          const float* __restrict__ a_dst, float* es, float* ed)
{
    int i = blockIdx.x * blockDim.x + threadIdx.x;
    if (i >= N * HEADS) return;
    int n = i >> 1, hh = i & 1;
    const float* row = h + (size_t)n * 256 + hh * HD;
    const float* as  = a_src + hh * HD;
    const float* ad  = a_dst + hh * HD;
    float s0 = 0.f, s1 = 0.f;
    for (int d = 0; d < HD; ++d) { float v = row[d]; s0 += v * as[d]; s1 += v * ad[d]; }
    es[i] = s0; ed[i] = s1;
}

// ---------------- per-node attention logits, layer 2 ----------------
__global__ void k_logits2(int N, const float* __restrict__ h, const float* __restrict__ a_src,
                          const float* __restrict__ a_dst, float* es, float* ed)
{
    int n = blockIdx.x * blockDim.x + threadIdx.x;
    if (n >= N) return;
    float s0 = 0.f, s1 = 0.f;
#pragma unroll
    for (int c = 0; c < COUT; ++c) { float v = h[(size_t)n * COUT + c]; s0 += v * a_src[c]; s1 += v * a_dst[c]; }
    es[n] = s0; ed[n] = s1;
}

static __device__ __forceinline__ void edge_nodes(const long long* ei, int E, int e, int& s, int& d)
{
    if (e < E) { s = (int)ei[e]; d = (int)ei[E + e]; }
    else       { s = d = e - E; }
}

// ---------------- segment max (leaky-relu logits), layer 1 ----------------
__global__ void k_edge_max1(const long long* ei, int E, int ET,
                            const float* es, const float* ed, unsigned* amax)
{
    int e = blockIdx.x * blockDim.x + threadIdx.x;
    if (e >= ET) return;
    int s, d; edge_nodes(ei, E, e, s, d);
#pragma unroll
    for (int h = 0; h < HEADS; ++h) {
        float a = es[s * HEADS + h] + ed[d * HEADS + h];
        a = a > 0.f ? a : 0.2f * a;
        atomicMax(&amax[d * HEADS + h], encf(a));
    }
}

// ---------------- exp + denom, layer 1 ----------------
__global__ void k_edge_exp1(const long long* ei, int E, int ET, const float* es, const float* ed,
                            const unsigned* amax, float* den, float* ex)
{
    int e = blockIdx.x * blockDim.x + threadIdx.x;
    if (e >= ET) return;
    int s, d; edge_nodes(ei, E, e, s, d);
#pragma unroll
    for (int h = 0; h < HEADS; ++h) {
        float a = es[s * HEADS + h] + ed[d * HEADS + h];
        a = a > 0.f ? a : 0.2f * a;
        float x = __expf(a - decf(amax[d * HEADS + h]));
        ex[(size_t)e * HEADS + h] = x;
        atomicAdd(&den[d * HEADS + h], x);
    }
}

// ---------------- weighted aggregation, layer 1 (wave per edge, 256 feats) ----------------
__global__ void k_aggregate1(const long long* ei, int E, int ET,
                             const float* __restrict__ h1, const float* __restrict__ ex,
                             const float* __restrict__ den, float* __restrict__ out1)
{
    int w    = (int)(((long long)blockIdx.x * blockDim.x + threadIdx.x) >> 5);
    int lane = threadIdx.x & 31;
    if (w >= ET) return;
    int s, d; edge_nodes(ei, E, w, s, d);
    int h = lane >> 4;   // lane*8 spans cols [lane*8, lane*8+8) -> head = lane/16
    float coeff = ex[(size_t)w * HEADS + h] / (den[d * HEADS + h] + 1e-16f);
    const float* sp = h1 + (size_t)s * 256 + lane * 8;
    float* dp = out1 + (size_t)d * 256 + lane * 8;
#pragma unroll
    for (int j = 0; j < 8; ++j) atomicAdd(&dp[j], sp[j] * coeff);
}

// ---------------- bias + ELU (in place) ----------------
__global__ void k_elu_bias(int N, float* out1, const float* __restrict__ b1)
{
    long long i = (long long)blockIdx.x * blockDim.x + threadIdx.x;
    if (i >= (long long)N * 256) return;
    float v = out1[i] + b1[i & 255];
    out1[i] = v > 0.f ? v : expm1f(v);
}

// ---------------- layer-2 segment max / exp / aggregate (1 head, 4 cols) ----------------
__global__ void k_edge_max2(const long long* ei, int E, int ET,
                            const float* es, const float* ed, unsigned* amax)
{
    int e = blockIdx.x * blockDim.x + threadIdx.x;
    if (e >= ET) return;
    int s, d; edge_nodes(ei, E, e, s, d);
    float a = es[s] + ed[d];
    a = a > 0.f ? a : 0.2f * a;
    atomicMax(&amax[d], encf(a));
}

__global__ void k_edge_exp2(const long long* ei, int E, int ET, const float* es, const float* ed,
                            const unsigned* amax, float* den, float* ex)
{
    int e = blockIdx.x * blockDim.x + threadIdx.x;
    if (e >= ET) return;
    int s, d; edge_nodes(ei, E, e, s, d);
    float a = es[s] + ed[d];
    a = a > 0.f ? a : 0.2f * a;
    float x = __expf(a - decf(amax[d]));
    ex[e] = x;
    atomicAdd(&den[d], x);
}

__global__ void k_aggregate2(const long long* ei, int E, int ET,
                             const float* __restrict__ h2, const float* __restrict__ ex,
                             const float* __restrict__ den, float* __restrict__ out)
{
    int e = blockIdx.x * blockDim.x + threadIdx.x;
    if (e >= ET) return;
    int s, d; edge_nodes(ei, E, e, s, d);
    float coeff = ex[e] / (den[d] + 1e-16f);
#pragma unroll
    for (int c = 0; c < COUT; ++c)
        atomicAdd(&out[(size_t)d * COUT + c], h2[(size_t)s * COUT + c] * coeff);
}

// =========================================================================================
extern "C" void kernel_launch(void* const* d_in, const int* in_sizes, int n_in,
                              void* d_out, int out_size, void* d_ws, size_t ws_size,
                              hipStream_t stream)
{
    const float*     x    = (const float*)d_in[0];
    const long long* ei   = (const long long*)d_in[1];
    const float*     W1   = (const float*)d_in[2];
    const float*     a1s  = (const float*)d_in[3];
    const float*     a1d  = (const float*)d_in[4];
    const float*     b1   = (const float*)d_in[5];
    const float*     W2   = (const float*)d_in[6];
    const float*     a2s  = (const float*)d_in[7];
    const float*     a2d  = (const float*)d_in[8];
    const float*     b2   = (const float*)d_in[9];
    float*           outp = (float*)d_out;

    const int N  = in_sizes[0] / 256;
    const int E  = in_sizes[1] / 2;
    const int ET = E + N;

    // -------- workspace layout --------
    size_t off = 0;
    auto alloc = [&](size_t bytes) -> void* {
        void* p = (char*)d_ws + off;
        off = (off + bytes + 255) & ~(size_t)255;
        return p;
    };
    unsigned short* Bt1  = (unsigned short*)alloc((size_t)256 * 256 * 2);
    unsigned short* Bt2  = (unsigned short*)alloc((size_t)16 * 256 * 2);
    float*          h1   = (float*)alloc((size_t)N * 256 * 4);
    float*          out1 = (float*)alloc((size_t)N * 256 * 4);
    float*          es1  = (float*)alloc((size_t)N * HEADS * 4);
    float*          ed1  = (float*)alloc((size_t)N * HEADS * 4);
    unsigned*       am1  = (unsigned*)alloc((size_t)N * HEADS * 4);
    float*          dn1  = (float*)alloc((size_t)N * HEADS * 4);
    float*          ex1  = (float*)alloc((size_t)ET * HEADS * 4);
    float*          h2   = (float*)alloc((size_t)N * COUT * 4);
    float*          es2  = (float*)alloc((size_t)N * 4);
    float*          ed2  = (float*)alloc((size_t)N * 4);
    unsigned*       am2  = (unsigned*)alloc((size_t)N * 4);
    float*          dn2  = (float*)alloc((size_t)N * 4);
    float*          ex2  = (float*)alloc((size_t)ET * 4);
    (void)ws_size; (void)n_in; (void)out_size;

    const int B = 256;
    const int M16 = (N + 15) / 16;

    // weights -> f16 transposed
    k_convert_w<<<(256 * 256 + 16 * 256 + B - 1) / B, B, 0, stream>>>(W1, W2, Bt1, Bt2);
    // accumulator init
    k_init<<<((long long)N * 256 + B - 1) / B, B, 0, stream>>>(N, out1, am1, dn1, am2, dn2, outp, b2);

    // ---- layer 1 ----
    k_gemm_wmma<<<dim3(M16, 2), B, 0, stream>>>(x, (const unsigned*)Bt1, h1, N, 16, 256, 256);
    k_logits1<<<(N * HEADS + B - 1) / B, B, 0, stream>>>(N, h1, a1s, a1d, es1, ed1);
    k_edge_max1<<<(ET + B - 1) / B, B, 0, stream>>>(ei, E, ET, es1, ed1, am1);
    k_edge_exp1<<<(ET + B - 1) / B, B, 0, stream>>>(ei, E, ET, es1, ed1, am1, dn1, ex1);
    k_aggregate1<<<(ET + 7) / 8, B, 0, stream>>>(ei, E, ET, h1, ex1, dn1, out1);
    k_elu_bias<<<((long long)N * 256 + B - 1) / B, B, 0, stream>>>(N, out1, b1);

    // ---- layer 2 ----
    k_gemm_wmma<<<dim3(M16, 1), B, 0, stream>>>(out1, (const unsigned*)Bt2, h2, N, 1, COUT, COUT);
    k_logits2<<<(N + B - 1) / B, B, 0, stream>>>(N, h2, a2s, a2d, es2, ed2);
    k_edge_max2<<<(ET + B - 1) / B, B, 0, stream>>>(ei, E, ET, es2, ed2, am2);
    k_edge_exp2<<<(ET + B - 1) / B, B, 0, stream>>>(ei, E, ET, es2, ed2, am2, dn2, ex2);
    k_aggregate2<<<(ET + B - 1) / B, B, 0, stream>>>(ei, E, ET, h2, ex2, dn2, outp);
}
